// AdaptivePoolCompressor_44375602103046
// MI455X (gfx1250) — matrix operand order repaired and checked by reading, not compiled, not run on hardware
//
#include <hip/hip_runtime.h>
#include <hip/hip_bf16.h>

// Problem constants (match reference)
#define B_ 8
#define S_ 8192
#define D_ 1024
#define T_ 512
#define H_ 256

typedef __attribute__((ext_vector_type(16))) _Float16 v16h;
typedef __attribute__((ext_vector_type(8)))  _Float16 v8h;
typedef __attribute__((ext_vector_type(8)))  float    v8f;
typedef __attribute__((ext_vector_type(4)))  unsigned int su4;
typedef __attribute__((ext_vector_type(8)))  unsigned int su8;

union AFrag { v16h v; v8h h[2]; };
union Acc8  { v8f  v; float f[8]; };
union H8    { v8h  v; _Float16 f[8]; };

// ---------------------------------------------------------------------------
// Tensor Data Mover: 2D tile between LDS and global (D# groups 0+1, §8.3/8.4).
// data_size code: 0=1B,1=2B,2=4B,3=8B. Dims/strides in data_size units.
// ---------------------------------------------------------------------------
__device__ __forceinline__ void tdm_2d(unsigned lds_off, const void* gaddr,
                                       unsigned ds_code,
                                       unsigned tile_d0, unsigned tile_d1,
                                       unsigned tensor_d0, unsigned tensor_d1,
                                       unsigned long long stride0, bool store) {
  unsigned long long ga = (unsigned long long)gaddr;
  su4 g0 = {
      1u,                                                   // count=1, user desc
      lds_off,                                              // lds_addr (bytes)
      (unsigned)ga,                                         // global_addr[31:0]
      ((unsigned)(ga >> 32) & 0x01FFFFFFu) | 0x80000000u    // ga[56:32] | type=2
  };
  su8 g1 = {
      ds_code << 16,                                        // mask=0, data_size
      (tensor_d0 & 0xFFFFu) << 16,                          // tensor_dim0[15:0]
      ((tensor_d0 >> 16) & 0xFFFFu) | ((tensor_d1 & 0xFFFFu) << 16),
      ((tensor_d1 >> 16) & 0xFFFFu) | (tile_d0 << 16),      // tile_dim0
      tile_d1,                                              // tile_dim1 (tile_dim2=0)
      (unsigned)stride0,                                    // dim0 stride [31:0]
      (unsigned)((stride0 >> 32) & 0xFFFFu),                // dim0 stride [47:32]
      0u
  };
  if (store)
    asm volatile("tensor_store_from_lds %0, %1" :: "s"(g0), "s"(g1) : "memory");
  else
    asm volatile("tensor_load_to_lds %0, %1" :: "s"(g0), "s"(g1) : "memory");
}

// ---------------------------------------------------------------------------
// Kernel 0: W1 [D,H] fp32 -> W1t [H,D] f16 (so WMMA B-fragments load contiguous)
// ---------------------------------------------------------------------------
__global__ __launch_bounds__(256) void k_w1_to_f16(const float* __restrict__ W1,
                                                   _Float16* __restrict__ W1t) {
  int idx = blockIdx.x * 256 + threadIdx.x;   // 0 .. H*D-1
  int h = idx >> 10;                          // / D_
  int d = idx & (D_ - 1);
  W1t[idx] = (_Float16)W1[d * H_ + h];
}

// ---------------------------------------------------------------------------
// Kernel 1: scores[b,s] = W2 . GELU(x[b,s,:] @ W1 + b1) + b2
// One block = 16 rows of x. 8 waves, each owns 2 N-tiles (16 cols) of H=256.
// ---------------------------------------------------------------------------
__global__ __launch_bounds__(256) void k_scores(
    const float* __restrict__ x, const _Float16* __restrict__ W1t,
    const float* __restrict__ b1, const float* __restrict__ W2,
    const float* __restrict__ b2, float* __restrict__ scores) {
  __shared__ alignas(16) _Float16 As[16 * 32];
  __shared__ float sums[16];

  const int tid  = threadIdx.x;
  const int wave = tid >> 5;
  const int lane = tid & 31;
  const int l16  = lane & 15;
  const int lhi  = lane >> 4;
  const size_t row0 = (size_t)blockIdx.x * 16;

  if (tid < 16) sums[tid] = 0.0f;

  v8f acc[2] = {};

  const int arow = tid >> 4;
  const int ak   = (tid & 15) * 2;
  const float* xrow = x + (row0 + (size_t)arow) * D_;

  for (int k0 = 0; k0 < D_; k0 += 32) {
    __syncthreads();
    float2 xv = *(const float2*)(xrow + k0 + ak);
    As[arow * 32 + ak]     = (_Float16)xv.x;
    As[arow * 32 + ak + 1] = (_Float16)xv.y;
    __syncthreads();

    AFrag a;
    const v8h* ap = (const v8h*)(As + l16 * 32 + lhi * 16);
    a.h[0] = ap[0];
    a.h[1] = ap[1];

    #pragma unroll
    for (int bt = 0; bt < 2; ++bt) {
      const int col = wave * 32 + bt * 16 + l16;
      const v8h* bp = (const v8h*)(W1t + (size_t)col * D_ + k0 + lhi * 16);
      AFrag bf;
      bf.h[0] = bp[0];
      bf.h[1] = bp[1];
      acc[bt] = __builtin_amdgcn_wmma_f32_16x16x32_f16(
          false, a.v, false, bf.v, (short)0, acc[bt], false, false);
    }
  }

  #pragma unroll
  for (int bt = 0; bt < 2; ++bt) {
    Acc8 u; u.v = acc[bt];
    const int col = wave * 32 + bt * 16 + l16;
    const float b1c = b1[col];
    const float w2c = W2[col];
    #pragma unroll
    for (int v = 0; v < 8; ++v) {
      const int row = v + 8 * lhi;
      const float val = u.f[v] + b1c;
      const float g = 0.5f * val * (1.0f + erff(val * 0.70710678118654752f));
      atomicAdd(&sums[row], g * w2c);
    }
  }
  __syncthreads();
  if (tid < 16) scores[row0 + tid] = sums[tid] + b2[0];
}

// ---------------------------------------------------------------------------
// Kernel 2: weights[b,:] = softmax(scores[b,:]) over S
// ---------------------------------------------------------------------------
__global__ __launch_bounds__(256) void k_softmax_w(const float* __restrict__ scores,
                                                   float* __restrict__ weights) {
  __shared__ float red[256];
  const int b = blockIdx.x, tid = threadIdx.x;
  const float* sc = scores + (size_t)b * S_;

  float m = -1e30f;
  for (int s = tid; s < S_; s += 256) m = fmaxf(m, sc[s]);
  red[tid] = m; __syncthreads();
  for (int off = 128; off > 0; off >>= 1) {
    if (tid < off) red[tid] = fmaxf(red[tid], red[tid + off]);
    __syncthreads();
  }
  m = red[0]; __syncthreads();

  float l = 0.0f;
  for (int s = tid; s < S_; s += 256) l += __expf(sc[s] - m);
  red[tid] = l; __syncthreads();
  for (int off = 128; off > 0; off >>= 1) {
    if (tid < off) red[tid] += red[tid + off];
    __syncthreads();
  }
  const float inv = 1.0f / red[0];

  float* w = weights + (size_t)b * S_;
  for (int s = tid; s < S_; s += 256) w[s] = __expf(sc[s] - m) * inv;
}

// ---------------------------------------------------------------------------
// Kernel 3: per (b,t): row max m and 1/sum(exp(c-m)) of
//           c[s] = -|pp[t] - s/(S-1)|*S + 10*weights[b,s]
// ---------------------------------------------------------------------------
__global__ __launch_bounds__(256) void k_rowstats(const float* __restrict__ weights,
                                                  const float* __restrict__ pp,
                                                  float* __restrict__ mArr,
                                                  float* __restrict__ invlArr) {
  __shared__ float red[256];
  const int t = blockIdx.x, b = blockIdx.y, tid = threadIdx.x;
  const float* w = weights + (size_t)b * S_;
  const float ppt = pp[t];
  const float invS1 = 1.0f / (float)(S_ - 1);
  const float Sf = (float)S_;

  float m = -1e30f;
  for (int s = tid; s < S_; s += 256) {
    float c = 10.0f * w[s] - fabsf(ppt - s * invS1) * Sf;
    m = fmaxf(m, c);
  }
  red[tid] = m; __syncthreads();
  for (int off = 128; off > 0; off >>= 1) {
    if (tid < off) red[tid] = fmaxf(red[tid], red[tid + off]);
    __syncthreads();
  }
  m = red[0]; __syncthreads();

  float l = 0.0f;
  for (int s = tid; s < S_; s += 256) {
    float c = 10.0f * w[s] - fabsf(ppt - s * invS1) * Sf;
    l += __expf(c - m);
  }
  red[tid] = l; __syncthreads();
  for (int off = 128; off > 0; off >>= 1) {
    if (tid < off) red[tid] += red[tid + off];
    __syncthreads();
  }
  if (tid == 0) {
    mArr[(size_t)b * T_ + t]    = m;
    invlArr[(size_t)b * T_ + t] = 1.0f / red[0];
  }
}

// ---------------------------------------------------------------------------
// Kernel 4: fused softmax-weights x GEMM:  out[b,t,:] = sum_s p[t,s] * x[b,s,:]
// Block: 256 threads (8 waves); tile = 128 t-rows x 128 d-cols (x re-read only
// T/128 = 4x). Weights row DMA'd to LDS once via TENSOR_LOAD_TO_LDS; output
// tile written to LDS and DMA'd out via TENSOR_STORE_FROM_LDS.
// ---------------------------------------------------------------------------
__global__ __launch_bounds__(256) void k_compress(
    const float* __restrict__ x, const float* __restrict__ weights,
    const float* __restrict__ pp, const float* __restrict__ mArr,
    const float* __restrict__ invlArr, float* __restrict__ out) {
  __shared__ union {
    struct {
      alignas(16) _Float16 As[128 * 32];   // [t-row][k]    8 KB
      alignas(16) _Float16 Bs[128 * 32];   // [d-local][k]  8 KB
      alignas(16) float    wAll[S_];       // weights row  32 KB
    } c;
    float outT[128 * 128];                 // output tile  64 KB
  } sm;

  const int tid  = threadIdx.x;
  const int wave = tid >> 5;
  const int lane = tid & 31;
  const int l16  = lane & 15;
  const int lhi  = lane >> 4;

  const int d0 = blockIdx.x * 128;
  const int t0 = blockIdx.y * 128;
  const int b  = blockIdx.z;

  const float* xB = x + (size_t)b * S_ * D_;

  // ---- TDM: whole weights row [S] fp32 -> LDS (wave 0 issues) ----
  if (wave == 0) {
    tdm_2d((unsigned)(unsigned long long)(uintptr_t)&sm.c.wAll[0],
           weights + (size_t)b * S_, /*4B*/2,
           /*tile*/S_, 1, /*tensor*/S_, 1, /*stride*/S_, /*store=*/false);
  }

  // per-thread A-gen constants (t-row fixed for whole K loop)
  const int arow = tid >> 1;            // 0..127
  const int ak0  = (tid & 1) * 16;      // 0 or 16
  const int tg   = t0 + arow;
  const float ppt  = pp[tg];
  const float mrow = mArr[(size_t)b * T_ + tg];
  const float invl = invlArr[(size_t)b * T_ + tg];
  const float invS1 = 1.0f / (float)(S_ - 1);
  const float Sf = (float)S_;

  const int xr = tid >> 5;              // 0..7 s sub-row per pass
  const int xc = (tid & 31) * 4;        // d column (float4 granular)

  v8f acc[8] = {};

  if (wave == 0) __builtin_amdgcn_s_wait_tensorcnt(0);
  __syncthreads();                      // wAll visible to all waves

  for (int s0 = 0; s0 < S_; s0 += 32) {
    __syncthreads();

    // ---- A tile: probabilities -> f16 (two b128 LDS stores per thread) ----
    #pragma unroll
    for (int q = 0; q < 2; ++q) {
      H8 pv;
      #pragma unroll
      for (int i = 0; i < 8; ++i) {
        const int s = s0 + ak0 + q * 8 + i;
        const float c = 10.0f * sm.c.wAll[s] - fabsf(ppt - s * invS1) * Sf;
        pv.f[i] = (_Float16)(__expf(c - mrow) * invl);
      }
      *(v8h*)(sm.c.As + arow * 32 + ak0 + q * 8) = pv.v;
    }

    // ---- B tile: x[s0..s0+31][d0..d0+127] -> f16 LDS transposed [d][k] ----
    #pragma unroll
    for (int p = 0; p < 4; ++p) {
      const int srow = p * 8 + xr;
      const float4 xv = *(const float4*)(xB + (size_t)(s0 + srow) * D_ + d0 + xc);
      sm.c.Bs[(xc + 0) * 32 + srow] = (_Float16)xv.x;
      sm.c.Bs[(xc + 1) * 32 + srow] = (_Float16)xv.y;
      sm.c.Bs[(xc + 2) * 32 + srow] = (_Float16)xv.z;
      sm.c.Bs[(xc + 3) * 32 + srow] = (_Float16)xv.w;
      if (p == 0 && s0 + 32 < S_)   // prefetch next x tile
        __builtin_prefetch(xB + (size_t)(s0 + 32 + srow) * D_ + d0 + xc, 0, 0);
    }
    __syncthreads();

    // ---- B fragment (this wave's 16-wide d tile), 8 WMMAs over t tiles ----
    AFrag bf;
    {
      const v8h* bp = (const v8h*)(sm.c.Bs + (wave * 16 + l16) * 32 + lhi * 16);
      bf.h[0] = bp[0];
      bf.h[1] = bp[1];
    }
    #pragma unroll
    for (int ti = 0; ti < 8; ++ti) {
      AFrag a;
      const v8h* ap = (const v8h*)(sm.c.As + (ti * 16 + l16) * 32 + lhi * 16);
      a.h[0] = ap[0];
      a.h[1] = ap[1];
      acc[ti] = __builtin_amdgcn_wmma_f32_16x16x32_f16(
          false, a.v, false, bf.v, (short)0, acc[ti], false, false);
    }
  }

  // ---- epilogue: deposit 128x128 fp32 tile in LDS, DMA out via TDM ----
  __syncthreads();   // all fragment reads of As/Bs done before union reuse
  #pragma unroll
  for (int ti = 0; ti < 8; ++ti) {
    Acc8 u; u.v = acc[ti];
    const int d = wave * 16 + l16;
    #pragma unroll
    for (int v = 0; v < 8; ++v) {
      const int t = ti * 16 + v + 8 * lhi;
      sm.outT[t * 128 + d] = u.f[v];
    }
  }
  __syncthreads();
  if (wave == 0) {
    tdm_2d((unsigned)(unsigned long long)(uintptr_t)&sm.outT[0],
           out + ((size_t)b * T_ + t0) * D_ + d0, /*4B*/2,
           /*tile*/128, 128, /*tensor*/D_, 128, /*stride*/D_, /*store=*/true);
    __builtin_amdgcn_s_wait_tensorcnt(0);
  }
}

// ---------------------------------------------------------------------------
// Host launch
// ---------------------------------------------------------------------------
extern "C" void kernel_launch(void* const* d_in, const int* in_sizes, int n_in,
                              void* d_out, int out_size, void* d_ws, size_t ws_size,
                              hipStream_t stream) {
  (void)in_sizes; (void)n_in; (void)out_size; (void)ws_size;
  const float* x  = (const float*)d_in[0];
  const float* W1 = (const float*)d_in[1];
  const float* b1 = (const float*)d_in[2];
  const float* W2 = (const float*)d_in[3];
  const float* b2 = (const float*)d_in[4];
  const float* pp = (const float*)d_in[5];
  float* out = (float*)d_out;

  // workspace layout (~1.06 MB total)
  char* ws = (char*)d_ws;
  _Float16* W1t    = (_Float16*)ws;                           // 512 KB
  float*    scores = (float*)(ws + (size_t)512 * 1024);       // 256 KB
  float*    wts    = (float*)(ws + (size_t)768 * 1024);       // 256 KB
  float*    mArr   = (float*)(ws + (size_t)1024 * 1024);      // 16 KB
  float*    invl   = (float*)(ws + (size_t)1040 * 1024);      // 16 KB

  k_w1_to_f16<<<(H_ * D_) / 256, 256, 0, stream>>>(W1, W1t);
  k_scores<<<(B_ * S_) / 16, 256, 0, stream>>>(x, W1t, b1, W2, b2, scores);
  k_softmax_w<<<B_, 256, 0, stream>>>(scores, wts);
  k_rowstats<<<dim3(T_, B_), 256, 0, stream>>>(wts, pp, mArr, invl);
  k_compress<<<dim3(D_ / 128, T_ / 128, B_), 256, 0, stream>>>(x, wts, pp, mArr, invl, out);
}
